// FFM_4372276707424
// MI455X (gfx1250) — matrix operand order repaired and checked by reading, not compiled, not run on hardware
//
#include <hip/hip_runtime.h>
#include <cstdint>
#include <cstddef>

typedef __attribute__((ext_vector_type(16))) _Float16 v16h;
typedef __attribute__((ext_vector_type(8)))  float    v8f;

#define F_DIM 2668
#define NFRAG 780   // 174 emb chunks * 4 N-tiles + 84 linear chunks * 1 tile

// Embedding order matches d_in[1..12]: au,ai,gu,gi,ou,oi,mu,mi,uu,ui,tu,ti, then lin (j=12)
static __device__ const int d_FRAGBASE[14] = {0,4,8,12,16,20,24,28,32,152,272,484,696,780};
static __device__ const int d_SSTART[13]   = {2626,2626,2626,2626,2628,2628,2649,2649,0,0,943,943,0};
static __device__ const int d_K[13]        = {1,1,2,2,21,21,19,19,943,943,1682,1682,2668};
static __device__ const int d_NCH[13]      = {1,1,1,1,1,1,1,1,30,30,53,53,84};
// 16B-aligned chunk anchors (weights zero-padded so fv reads stay in-row & results exact)
static __device__ const int d_ALIGN[13]    = {2624,2624,2624,2624,2628,2628,2636,2636,0,0,940,940,0};
static __device__ const int d_LASTCOL[13]  = {2624,2624,2624,2624,2628,2628,2636,2636,928,928,2604,2604,2636};
// last-chunk responsibility floor (only linear's end-anchored tail overlaps its neighbor)
static __device__ const int d_LOLAST[13]   = {2624,2624,2624,2624,2628,2628,2636,2636,928,928,2604,2604,2656};

// ---------------------------------------------------------------------------
// Prep: pack f32 weights into f16 WMMA B-fragment layout (32x16 K-major tiles).
// B layout (16-bit, wave32): lane n = col N (n=lane&15); lanes0-15 hold K 0..7,16..23,
// lanes16-31 hold K 8..15,24..31; VGPR v holds K pair lo/hi halves.
// Fragment f: 1024B = 32 lanes * 8 dwords. One thread packs one dword.
// ---------------------------------------------------------------------------
__global__ void pack_w_kernel(const float* w0, const float* w1, const float* w2, const float* w3,
                              const float* w4, const float* w5, const float* w6, const float* w7,
                              const float* w8, const float* w9, const float* w10, const float* w11,
                              const float* linw, uint32_t* __restrict__ pk) {
  const int id     = (int)blockIdx.x * 256 + (int)threadIdx.x;
  const int frag   = id >> 8;
  const int within = id & 255;
  const int lane   = within >> 3;
  const int v      = within & 7;

  int j = 0;
  while (j < 12 && frag >= d_FRAGBASE[j + 1]) ++j;
  const int rel = frag - d_FRAGBASE[j];
  const int nt  = (j == 12) ? 1 : 4;
  const int c   = rel / nt;
  const int t   = rel - c * nt;

  const int sstart   = d_SSTART[j];
  const int nch      = d_NCH[j];
  const int colstart = (c == nch - 1) ? d_LASTCOL[j] : (d_ALIGN[j] + 32 * c);
  int lo             = (c == nch - 1) ? d_LOLAST[j]  : colstart;
  if (lo < sstart) lo = sstart;
  const int hi       = sstart + d_K[j];

  const int n    = lane & 15;
  const int kic0 = ((lane >> 4) << 3) + 2 * v + ((v >> 2) << 3);   // B layout K within chunk

  const float* Wsrc[13] = {w0,w1,w2,w3,w4,w5,w6,w7,w8,w9,w10,w11,linw};
  const float* src = Wsrc[j];

  uint32_t packed = 0;
  #pragma unroll
  for (int h = 0; h < 2; ++h) {
    const int wrow = colstart + kic0 + h;        // fv column this K position reads
    float val = 0.0f;
    if (wrow >= lo && wrow < hi) {
      const int r = wrow - sstart;               // row within the weight matrix
      if (j < 12) val = src[r * 64 + t * 16 + n];
      else        val = (n == 0) ? src[r] : 0.0f;   // lin_w as column 0 of a 16-wide tile
    }
    const _Float16 hf = (_Float16)val;
    uint16_t bits;
    __builtin_memcpy(&bits, &hf, sizeof(bits));
    packed |= ((uint32_t)bits) << (16 * h);
  }
  pk[frag * 256 + lane * 8 + v] = packed;
}

// ---------------------------------------------------------------------------
// A fragment (16x32 f16, wave32): lane m = row (lane&15); kbase = (lane>>4)*8;
// VGPRs 0..3 hold K = kbase..kbase+7 (pairs), VGPRs 4..7 hold K = 16+kbase..+7.
// colstart is always a multiple of 4 floats -> 16B-aligned b128 loads.
// ---------------------------------------------------------------------------
__device__ __forceinline__ v16h load_a(const float* __restrict__ fvrow, int colstart, int kbase) {
  const float4* p0 = (const float4*)(fvrow + colstart + kbase);
  const float4* p1 = (const float4*)(fvrow + colstart + 16 + kbase);
  const float4 a0 = p0[0], a1 = p0[1];
  const float4 b0 = p1[0], b1 = p1[1];
  v16h a;
  a[0]  = (_Float16)a0.x; a[1]  = (_Float16)a0.y; a[2]  = (_Float16)a0.z; a[3]  = (_Float16)a0.w;
  a[4]  = (_Float16)a1.x; a[5]  = (_Float16)a1.y; a[6]  = (_Float16)a1.z; a[7]  = (_Float16)a1.w;
  a[8]  = (_Float16)b0.x; a[9]  = (_Float16)b0.y; a[10] = (_Float16)b0.z; a[11] = (_Float16)b0.w;
  a[12] = (_Float16)b1.x; a[13] = (_Float16)b1.y; a[14] = (_Float16)b1.z; a[15] = (_Float16)b1.w;
  return a;
}

#define WMMA(ACC, A, B) \
  (ACC) = __builtin_amdgcn_wmma_f32_16x16x32_f16(false, (A), false, (B), (short)0, (ACC), false, false)

// Fused user/item pair GEMM: one A fragment feeds 8 WMMAs (4 N-tiles x 2 weights).
template <int NCH>
__device__ __forceinline__ void gemm_pair(const float* __restrict__ fvrow, int kbase, int lane,
                                          const v16h* __restrict__ pk, int align, int lastcol,
                                          int fbu, int fbi, v8f (&accu)[4], v8f (&acci)[4]) {
  for (int c = 0; c < NCH; ++c) {
    const int colstart = (c == NCH - 1) ? lastcol : (align + 32 * c);
    const v16h A = load_a(fvrow, colstart, kbase);
    #pragma unroll
    for (int t = 0; t < 4; ++t) {
      const v16h Bu = pk[(size_t)(fbu + c * 4 + t) * 32 + lane];
      WMMA(accu[t], A, Bu);
    }
    #pragma unroll
    for (int t = 0; t < 4; ++t) {
      const v16h Bi = pk[(size_t)(fbi + c * 4 + t) * 32 + lane];
      WMMA(acci[t], A, Bi);
    }
  }
}

// userid pair + fused linear chunks (identical 16B-aligned colstarts 0,32,...,928)
template <int NCH>
__device__ __forceinline__ void gemm_pair_lin(const float* __restrict__ fvrow, int kbase, int lane,
                                              const v16h* __restrict__ pk, int align, int lastcol,
                                              int fbu, int fbi, int fbl,
                                              v8f (&accu)[4], v8f (&acci)[4], v8f& lin) {
  for (int c = 0; c < NCH; ++c) {
    const int colstart = (c == NCH - 1) ? lastcol : (align + 32 * c);
    const v16h A = load_a(fvrow, colstart, kbase);
    #pragma unroll
    for (int t = 0; t < 4; ++t) {
      const v16h Bu = pk[(size_t)(fbu + c * 4 + t) * 32 + lane];
      WMMA(accu[t], A, Bu);
    }
    #pragma unroll
    for (int t = 0; t < 4; ++t) {
      const v16h Bi = pk[(size_t)(fbi + c * 4 + t) * 32 + lane];
      WMMA(acci[t], A, Bi);
    }
    const v16h Bl = pk[(size_t)(fbl + c) * 32 + lane];
    WMMA(lin, A, Bl);
  }
}

// ---------------------------------------------------------------------------
// Main: 4 waves/block, each wave owns a 16-row M tile. Cross terms grouped as
//   gu*au + ou*(au+gu) + uu*(au+gu+ou) + (mu+tu)*(ai+gi+oi+ui) + ti*mi
// so peak liveness is 21 v8f (cross, AU, AI, TU, TI, lin) -> no VGPR-MSB zone.
// ---------------------------------------------------------------------------
__global__ __launch_bounds__(128) void ffm_kernel(const float* __restrict__ fv,
                                                  const v16h* __restrict__ pk,
                                                  const float* __restrict__ lin_b,
                                                  float* __restrict__ out) {
  const int lane  = (int)(threadIdx.x & 31u);
  const int wave  = (int)(threadIdx.x >> 5u);
  const int row0  = ((int)blockIdx.x * 4 + wave) * 16;
  const float* fvrow = fv + (size_t)(row0 + (lane & 15)) * F_DIM;
  const int kbase = (lane >> 4) * 8;

  const v8f vzero = {};
  v8f lin = {};
  v8f cross[4] = {};
  v8f AU[4] = {}, AI[4] = {};   // running user/item sums
  v8f TU[4] = {}, TI[4] = {};   // temporaries
  v8f MI[4] = {};               // movie-item, kept for ti*mi

  // age: au, ai
  gemm_pair<1>(fvrow, kbase, lane, pk, 2624, 2624, 0, 4, AU, AI);
  // gender: gu, gi   -> cross += gu*au ; SU = au+gu ; SI = ai+gi
  gemm_pair<1>(fvrow, kbase, lane, pk, 2624, 2624, 8, 12, TU, TI);
  #pragma unroll
  for (int t = 0; t < 4; ++t) {
    cross[t] += TU[t] * AU[t]; AU[t] += TU[t]; AI[t] += TI[t];
    TU[t] = vzero; TI[t] = vzero;
  }
  // occupation: ou, oi -> cross += ou*(au+gu) ; SU += ou ; SI += oi
  gemm_pair<1>(fvrow, kbase, lane, pk, 2628, 2628, 16, 20, TU, TI);
  #pragma unroll
  for (int t = 0; t < 4; ++t) {
    cross[t] += TU[t] * AU[t]; AU[t] += TU[t]; AI[t] += TI[t];
    TU[t] = vzero; TI[t] = vzero;
  }
  // userid (+ linear chunks 0..29): cross += uu*SU ; SI += ui ; AU dead after
  gemm_pair_lin<30>(fvrow, kbase, lane, pk, 0, 928, 32, 152, 696, TU, TI, lin);
  #pragma unroll
  for (int t = 0; t < 4; ++t) {
    cross[t] += TU[t] * AU[t]; AI[t] += TI[t];
    TU[t] = vzero; TI[t] = vzero;
  }
  // movie: mu, mi -> cross += mu*(ai+gi+oi+ui) ; keep MI
  gemm_pair<1>(fvrow, kbase, lane, pk, 2636, 2636, 24, 28, TU, MI);
  #pragma unroll
  for (int t = 0; t < 4; ++t) { cross[t] += TU[t] * AI[t]; TU[t] = vzero; }
  // itemid: tu, ti -> cross += tu*(ai+gi+oi+ui) + ti*mi
  gemm_pair<53>(fvrow, kbase, lane, pk, 940, 2604, 272, 484, TU, TI);
  #pragma unroll
  for (int t = 0; t < 4; ++t) {
    cross[t] += TU[t] * AI[t]; cross[t] += TI[t] * MI[t];
  }
  // linear tail chunks 30..83 (cols 960..2667; last chunk end-anchored at 2636)
  for (int c = 30; c < 84; ++c) {
    const int colstart = (c == 83) ? 2636 : (32 * c);
    const v16h A  = load_a(fvrow, colstart, kbase);
    const v16h Bl = pk[(size_t)(696 + c) * 32 + lane];
    WMMA(lin, A, Bl);
  }

  // Reduce cross over D (C layout: lane holds fixed N, vgpr v = row m%8,
  // lanes 0-15 -> rows 0..7, lanes 16-31 -> rows 8..15). lin column 0 lives
  // on lanes 0 and 16 -- exactly the storing lanes.
  const float bias = lin_b[0];
  float res[8];
  #pragma unroll
  for (int v = 0; v < 8; ++v) {
    float cs = cross[0][v] + cross[1][v] + cross[2][v] + cross[3][v];
    cs += __shfl_xor(cs, 1, 32);
    cs += __shfl_xor(cs, 2, 32);
    cs += __shfl_xor(cs, 4, 32);
    cs += __shfl_xor(cs, 8, 32);
    const float logit = lin[v] + bias + cs;
    res[v] = 1.0f / (1.0f + __expf(-logit));
  }
  if ((lane & 15) == 0) {
    const int mb = (lane >> 4) * 8;
    #pragma unroll
    for (int v = 0; v < 8; ++v) out[row0 + mb + v] = res[v];
  }
}

extern "C" void kernel_launch(void* const* d_in, const int* in_sizes, int n_in,
                              void* d_out, int out_size, void* d_ws, size_t ws_size,
                              hipStream_t stream) {
  (void)in_sizes; (void)n_in; (void)out_size; (void)ws_size;
  const float* fv   = (const float*)d_in[0];
  const float* linw = (const float*)d_in[13];
  const float* linb = (const float*)d_in[14];
  uint32_t*    pk   = (uint32_t*)d_ws;          // 780 KiB packed-fragment buffer

  pack_w_kernel<<<dim3(NFRAG), dim3(256), 0, stream>>>(
      (const float*)d_in[1],  (const float*)d_in[2],  (const float*)d_in[3],
      (const float*)d_in[4],  (const float*)d_in[5],  (const float*)d_in[6],
      (const float*)d_in[7],  (const float*)d_in[8],  (const float*)d_in[9],
      (const float*)d_in[10], (const float*)d_in[11], (const float*)d_in[12],
      linw, pk);

  // 32768 rows / (4 waves * 16 rows) = 512 blocks
  ffm_kernel<<<dim3(512), dim3(128), 0, stream>>>(fv, (const v16h*)d_ws, linb, (float*)d_out);
}
